// SelfAttention_32770600469103
// MI455X (gfx1250) — matrix operand order repaired and checked by reading, not compiled, not run on hardware
//
#include <hip/hip_runtime.h>
#include <math.h>

// ---------------------------------------------------------------------------
// Self-attention forward for MI455X (gfx1250, wave32, WMMA, async-LDS copies).
//   B=4, S=2048, D_MODEL=2048, H=16, D_K=128, causal mask (runtime flag).
// Pipeline: f32->bf16 convert -> Q/K/V projection GEMMs (LDS double-buffered,
//           async global->LDS, WMMA bf16/f32-acc) -> flash attention
//           (WMMA scores + WMMA P@V, online softmax) -> output GEMM (f32).
// ---------------------------------------------------------------------------

typedef __attribute__((ext_vector_type(16))) __bf16 v16bf;
typedef __attribute__((ext_vector_type(8)))  __bf16 v8bf;
typedef __attribute__((ext_vector_type(4)))  __bf16 v4bf;
typedef __attribute__((ext_vector_type(8)))  float  v8f;

#define DEV_INLINE __device__ __forceinline__

constexpr int BATCH  = 4;
constexpr int SEQ    = 2048;
constexpr int DMODEL = 2048;
constexpr int NHEAD  = 16;
constexpr int DKH    = 128;                       // d_k per head
constexpr int KB     = 32;                        // GEMM k-step
constexpr int LDT    = 40;                        // padded LDS row stride
                                                  // (40*2B=80B -> 16 rows hit
                                                  //  all 64 banks once)
constexpr float QSCALE = 0.08838834764831845f;    // 1/sqrt(128)

DEV_INLINE v8f wmma_bf16(v16bf a, v16bf b, v8f c) {
  // D = A(16x32 bf16) * B(32x16 bf16) + C(16x16 f32)
  return __builtin_amdgcn_wmma_f32_16x16x32_bf16(
      false, a, false, b, (short)0, c, false, false);
}

union V16u { v16bf v; v8bf h[2]; };

// A-fragment (16x32 bf16) from a row-major source, leading dim `ld`.
// ISA 7.12.2: lane<16 : row=lane,    elems0..7=K[0..7],  elems8..15=K[16..23]
//             lane>=16: row=lane-16, elems0..7=K[8..15], elems8..15=K[24..31]
DEV_INLINE v16bf load_a_frag(const __bf16* base, int ld, int lane) {
  const __bf16* p = base + (lane & 15) * ld + ((lane >> 4) << 3);
  V16u u;
  u.h[0] = *(const v8bf*)(p);
  u.h[1] = *(const v8bf*)(p + 16);
  return u.v;
}

// B-fragment (32x16 bf16): column n = source row n, same K order as A-fragment:
// lane<16: col=lane, K=0..7,16..23 ; lane>=16: col=lane-16, K=8..15,24..31
DEV_INLINE v16bf load_b_frag(const __bf16* base, int ld, int lane) {
  const __bf16* p = base + (lane & 15) * ld + ((lane >> 4) << 3);
  V16u u;
  u.h[0] = *(const v8bf*)(p);
  u.h[1] = *(const v8bf*)(p + 16);
  return u.v;
}

// Async 16B global -> LDS copy (per-lane addresses, tracked by ASYNCcnt).
// Low 32 bits of a generic shared pointer are the LDS byte address (ISA 10.2).
DEV_INLINE void cp16_async(void* lds, const void* g) {
  unsigned int lo = (unsigned int)(size_t)lds;
  asm volatile("global_load_async_to_lds_b128 %0, %1, off"
               :: "v"(lo), "v"(g) : "memory");
}
DEV_INLINE void wait_async0() {
  asm volatile("s_wait_asynccnt 0x0" ::: "memory");
}

// ---------------------------------------------------------------------------
__global__ void __launch_bounds__(256)
cvt_f32_to_bf16(const float* __restrict__ in, __bf16* __restrict__ out, int n4) {
  int i = blockIdx.x * 256 + threadIdx.x;
  if (i < n4) {
    const float4 f = ((const float4*)in)[i];
    v4bf o;
    o[0] = (__bf16)f.x; o[1] = (__bf16)f.y;
    o[2] = (__bf16)f.z; o[3] = (__bf16)f.w;
    ((v4bf*)out)[i] = o;
  }
}

// ---------------------------------------------------------------------------
// Y[m,n] = sum_k A[m,k] * W[n,k]   (M=8192, N=2048, K=2048)
// Block tile 128x128, 8 waves (2x4), wave tile 64x32 (4x2 WMMA tiles).
// K streamed in 32-wide steps, double-buffered through LDS via async copies.
// MODE 0: Q -> [B,H,S,dk] bf16 (scaled)   MODE 1: K -> [B,H,S,dk] bf16
// MODE 2: V -> [B,H,dk,S] bf16 (transposed)   MODE 3: [M,N] f32 out
template <int MODE>
__global__ void __launch_bounds__(256)
gemm_bf16(const __bf16* __restrict__ A, const __bf16* __restrict__ W,
          __bf16* __restrict__ outb, float* __restrict__ outf) {
  __shared__ __bf16 As[2][128 * LDT];
  __shared__ __bf16 Bs[2][128 * LDT];

  const int tid  = threadIdx.x;
  const int lane = tid & 31;
  const int wave = tid >> 5;
  const int mb = blockIdx.y * 128;
  const int nb = blockIdx.x * 128;
  const int mw = (wave >> 2) * 64;            // wave M offset in block tile
  const int nw = (wave & 3) * 32;             // wave N offset in block tile

  const __bf16* Ag = A + (size_t)mb * DMODEL;
  const __bf16* Wg = W + (size_t)nb * DMODEL;

  // 512 16B-chunks per 128x32 tile; each thread stages 2 per tile per k-step.
  const int r0 = tid >> 2,          c0 = (tid & 3) * 8;           // rows 0..63
  const int r1 = (tid + 256) >> 2,  c1 = ((tid + 256) & 3) * 8;   // rows 64..127

  auto stage = [&](int k, int buf) {
    cp16_async(&As[buf][r0 * LDT + c0], Ag + (size_t)r0 * DMODEL + k + c0);
    cp16_async(&As[buf][r1 * LDT + c1], Ag + (size_t)r1 * DMODEL + k + c1);
    cp16_async(&Bs[buf][r0 * LDT + c0], Wg + (size_t)r0 * DMODEL + k + c0);
    cp16_async(&Bs[buf][r1 * LDT + c1], Wg + (size_t)r1 * DMODEL + k + c1);
  };

  v8f acc[4][2] = {};
  stage(0, 0);

  constexpr int KSTEPS = DMODEL / KB;         // 64
  for (int ks = 0; ks < KSTEPS; ++ks) {
    wait_async0();                            // our copies for buf[ks&1] done
    __syncthreads();                          // everyone's copies visible; all
                                              // reads of buf[(ks+1)&1] retired
    if (ks + 1 < KSTEPS) stage((ks + 1) * KB, (ks + 1) & 1);

    const __bf16* a = &As[ks & 1][0];
    const __bf16* b = &Bs[ks & 1][0];
    v16bf bf0 = load_b_frag(b + (nw +  0) * LDT, LDT, lane);
    v16bf bf1 = load_b_frag(b + (nw + 16) * LDT, LDT, lane);
#pragma unroll
    for (int mi = 0; mi < 4; ++mi) {
      v16bf af = load_a_frag(a + (mw + mi * 16) * LDT, LDT, lane);
      acc[mi][0] = wmma_bf16(af, bf0, acc[mi][0]);
      acc[mi][1] = wmma_bf16(af, bf1, acc[mi][1]);
    }
  }

  // C/D layout: VGPR r, lanes 0-15 -> (row r, col lane); lanes 16-31 -> row r+8.
  const int col = lane & 15;
  const int rb  = (lane >> 4) << 3;
#pragma unroll
  for (int mi = 0; mi < 4; ++mi)
#pragma unroll
    for (int ni = 0; ni < 2; ++ni)
#pragma unroll
      for (int r = 0; r < 8; ++r) {
        const int gm = mb + mw + mi * 16 + rb + r;   // token index
        const int gn = nb + nw + ni * 16 + col;      // d_model index
        float v = acc[mi][ni][r];
        if constexpr (MODE == 3) {
          outf[(size_t)gm * DMODEL + gn] = v;
        } else {
          const int bb = gm >> 11, ss = gm & (SEQ - 1);
          const int hh = gn >> 7,  dd = gn & (DKH - 1);
          if constexpr (MODE == 0) v *= QSCALE;
          size_t idx;
          if constexpr (MODE == 2)
            idx = ((size_t)(bb * NHEAD + hh) * DKH + dd) * SEQ + ss;   // V^T
          else
            idx = ((size_t)(bb * NHEAD + hh) * SEQ + ss) * DKH + dd;   // Q/K
          outb[idx] = (__bf16)v;
        }
      }
}

// ---------------------------------------------------------------------------
// Flash attention: 1 wave per 16-query tile, 8 waves/block (128 queries).
// grid.x = B*H * (S/128) = 1024.  Q pre-scaled by 1/sqrt(dk).
__global__ void __launch_bounds__(256)
attn_kernel(const __bf16* __restrict__ Q, const __bf16* __restrict__ K,
            const __bf16* __restrict__ Vt, __bf16* __restrict__ Oattn,
            const int* __restrict__ use_mask) {
  __shared__ __bf16 Pt[8][16 * LDT];            // per-wave P tile (padded)

  const int lane = threadIdx.x & 31;
  const int wave = threadIdx.x >> 5;
  const int bh   = blockIdx.x >> 4;             // S/128 = 16 q-blocks
  const int qblk = blockIdx.x & 15;
  const int q0   = qblk * 128 + wave * 16;
  const int b    = bh >> 4;                     // H = 16
  const int h    = bh & 15;
  const int um   = *use_mask;

  const __bf16* Qb = Q  + (size_t)bh * SEQ * DKH;
  const __bf16* Kb = K  + (size_t)bh * SEQ * DKH;
  const __bf16* Vb = Vt + (size_t)bh * DKH * SEQ;
  __bf16* Pw = &Pt[wave][0];

  v16bf qf[4];
#pragma unroll
  for (int c = 0; c < 4; ++c)
    qf[c] = load_a_frag(Qb + (size_t)q0 * DKH + c * 32, DKH, lane);

  v8f   o[8] = {};                              // 16x128 f32 O accumulator
  float m[8], l[8];
#pragma unroll
  for (int r = 0; r < 8; ++r) { m[r] = -INFINITY; l[r] = 0.0f; }

  const int col = lane & 15;
  const int rb  = (lane >> 4) << 3;
  const int nkb = um ? ((q0 + 16 + 31) >> 5) : (SEQ >> 5);

  for (int kb = 0; kb < nkb; ++kb) {
    const int n0 = kb * 32;

    v8f s0 = {}, s1 = {};
#pragma unroll
    for (int c = 0; c < 4; ++c) {
      v16bf k0 = load_b_frag(Kb + (size_t)n0 * DKH + c * 32, DKH, lane);
      v16bf k1 = load_b_frag(Kb + (size_t)(n0 + 16) * DKH + c * 32, DKH, lane);
      s0 = wmma_bf16(qf[c], k0, s0);
      s1 = wmma_bf16(qf[c], k1, s1);
    }

    float al[8];
#pragma unroll
    for (int r = 0; r < 8; ++r) {
      const int row = q0 + rb + r;
      float a0 = (!um || (n0 + col      <= row)) ? s0[r] : -INFINITY;
      float a1 = (!um || (n0 + 16 + col <= row)) ? s1[r] : -INFINITY;
      float mx = fmaxf(a0, a1);
#pragma unroll
      for (int off = 1; off < 16; off <<= 1)
        mx = fmaxf(mx, __shfl_xor(mx, off, 32));
      const float mn = fmaxf(m[r], mx);
      const float sc = __expf(m[r] - mn);
      const float p0 = __expf(a0 - mn);
      const float p1 = __expf(a1 - mn);
      float rs = p0 + p1;
#pragma unroll
      for (int off = 1; off < 16; off <<= 1)
        rs += __shfl_xor(rs, off, 32);
      l[r] = l[r] * sc + rs;
      m[r] = mn;
      al[r] = sc;
      Pw[(rb + r) * LDT + col]      = (__bf16)p0;
      Pw[(rb + r) * LDT + col + 16] = (__bf16)p1;
    }

#pragma unroll
    for (int n = 0; n < 8; ++n)
#pragma unroll
      for (int r = 0; r < 8; ++r)
        o[n][r] *= al[r];

    // Wave-private LDS tile: DS ops are in-order per wave; force the wait and
    // stop compiler reordering across the store->load edge.
    asm volatile("s_wait_dscnt 0x0" ::: "memory");
    v16bf pf = load_a_frag(Pw, LDT, lane);

#pragma unroll
    for (int n = 0; n < 8; ++n) {
      v16bf vf = load_b_frag(Vb + (size_t)(n * 16) * SEQ + n0, SEQ, lane);
      o[n] = wmma_bf16(pf, vf, o[n]);
    }
  }

#pragma unroll
  for (int r = 0; r < 8; ++r) l[r] = 1.0f / l[r];
  const size_t obase = ((size_t)b * SEQ + q0) * DMODEL + (size_t)h * DKH;
#pragma unroll
  for (int n = 0; n < 8; ++n)
#pragma unroll
    for (int r = 0; r < 8; ++r)
      Oattn[obase + (size_t)(rb + r) * DMODEL + n * 16 + col] =
          (__bf16)(o[n][r] * l[r]);
}

// ---------------------------------------------------------------------------
extern "C" void kernel_launch(void* const* d_in, const int* in_sizes, int n_in,
                              void* d_out, int out_size, void* d_ws, size_t ws_size,
                              hipStream_t stream) {
  (void)in_sizes; (void)n_in; (void)out_size; (void)ws_size;
  const float* x  = (const float*)d_in[0];
  const float* Wq = (const float*)d_in[1];
  const float* Wk = (const float*)d_in[2];
  const float* Wv = (const float*)d_in[3];
  const float* Wo = (const float*)d_in[4];
  const int* use_mask = (const int*)d_in[5];

  const size_t nx = (size_t)BATCH * SEQ * DMODEL;   // 16,777,216
  const size_t nw = (size_t)DMODEL * DMODEL;        //  4,194,304

  char* p = (char*)d_ws;                            // ~192 MB total
  __bf16* xb  = (__bf16*)p; p += nx * 2;
  __bf16* Wqb = (__bf16*)p; p += nw * 2;
  __bf16* Wkb = (__bf16*)p; p += nw * 2;
  __bf16* Wvb = (__bf16*)p; p += nw * 2;
  __bf16* Wob = (__bf16*)p; p += nw * 2;
  __bf16* Qh  = (__bf16*)p; p += nx * 2;            // [B,H,S,dk]
  __bf16* Kh  = (__bf16*)p; p += nx * 2;            // [B,H,S,dk]
  __bf16* Vt  = (__bf16*)p; p += nx * 2;            // [B,H,dk,S]
  __bf16* At  = (__bf16*)p; p += nx * 2;            // [B,S,D]

  cvt_f32_to_bf16<<<(int)(nx / 4 / 256), 256, 0, stream>>>(x,  xb,  (int)(nx / 4));
  cvt_f32_to_bf16<<<(int)(nw / 4 / 256), 256, 0, stream>>>(Wq, Wqb, (int)(nw / 4));
  cvt_f32_to_bf16<<<(int)(nw / 4 / 256), 256, 0, stream>>>(Wk, Wkb, (int)(nw / 4));
  cvt_f32_to_bf16<<<(int)(nw / 4 / 256), 256, 0, stream>>>(Wv, Wvb, (int)(nw / 4));
  cvt_f32_to_bf16<<<(int)(nw / 4 / 256), 256, 0, stream>>>(Wo, Wob, (int)(nw / 4));

  dim3 grid(DMODEL / 128, (BATCH * SEQ) / 128);     // (16, 64)
  gemm_bf16<0><<<grid, 256, 0, stream>>>(xb, Wqb, Qh, nullptr);
  gemm_bf16<1><<<grid, 256, 0, stream>>>(xb, Wkb, Kh, nullptr);
  gemm_bf16<2><<<grid, 256, 0, stream>>>(xb, Wvb, Vt, nullptr);

  attn_kernel<<<BATCH * NHEAD * (SEQ / 128), 256, 0, stream>>>(Qh, Kh, Vt, At, use_mask);

  gemm_bf16<3><<<grid, 256, 0, stream>>>(At, Wob, nullptr, (float*)d_out);
}
